// SCBAB_30356828848469
// MI455X (gfx1250) — matrix operand order repaired and checked by reading, not compiled, not run on hardware
//
#include <hip/hip_runtime.h>

typedef _Float16 v16h __attribute__((ext_vector_type(16)));
typedef _Float16 v8h  __attribute__((ext_vector_type(8)));
typedef float    v8f  __attribute__((ext_vector_type(8)));

// ---------------------------------------------------------------------------
// small prep kernels
// ---------------------------------------------------------------------------
__global__ __launch_bounds__(256)
void f2h_kernel(const float* __restrict__ s, _Float16* __restrict__ d, int n) {
  int i = blockIdx.x * 256 + threadIdx.x;
  if (i < n) d[i] = (_Float16)s[i];
}

// bias_full[h][r][c] = rpb[rpi[r*64+c]*6 + h]   (6 x 64 x 64)
__global__ __launch_bounds__(256)
void biasexp_kernel(const int* __restrict__ rpi, const float* __restrict__ rpb,
                    float* __restrict__ bf) {
  int i = blockIdx.x * 256 + threadIdx.x;
  if (i < 6 * 4096) {
    int h = i >> 12, rc = i & 4095;
    bf[i] = rpb[rpi[rc] * 6 + h];
  }
}

// ---------------------------------------------------------------------------
// LayerNorm: one wave per row of 192; mode 0 writes rows in shifted-window
// order (roll(-4,-4) + window partition fused into the output mapping),
// mode 1 is identity row order. Output f16.
// ---------------------------------------------------------------------------
__global__ __launch_bounds__(256)
void ln_kernel(const float* __restrict__ x, const float* __restrict__ g,
               const float* __restrict__ b, _Float16* __restrict__ out, int mode) {
  int lane = threadIdx.x & 31;
  int row  = blockIdx.x * 8 + (threadIdx.x >> 5);   // output row (window*64+tok or pixel)
  size_t src;
  if (mode == 0) {
    int win = row >> 6, tok = row & 63;
    int bb = win >> 10, wi = win & 1023;
    int hh = ((wi >> 5) << 3) + (tok >> 3);
    int ww = ((wi & 31) << 3) + (tok & 7);
    int sh = (hh + 4) & 255, sw = (ww + 4) & 255;   // roll(-SS): dest h <- src h+SS
    src = (size_t)bb * 65536 + (size_t)sh * 256 + sw;
  } else {
    src = (size_t)row;
  }
  const float* xr = x + src * 192;
  float v[6]; float s = 0.f;
#pragma unroll
  for (int t = 0; t < 6; t++) { v[t] = xr[lane + 32 * t]; s += v[t]; }
#pragma unroll
  for (int m = 16; m >= 1; m >>= 1) s += __shfl_xor(s, m, 32);
  float mean = s * (1.f / 192.f);
  float vs = 0.f;
#pragma unroll
  for (int t = 0; t < 6; t++) { float d = v[t] - mean; vs += d * d; }
#pragma unroll
  for (int m = 16; m >= 1; m >>= 1) vs += __shfl_xor(vs, m, 32);
  float inv = rsqrtf(vs * (1.f / 192.f) + 1e-5f);
  _Float16* orow = out + (size_t)row * 192;
#pragma unroll
  for (int t = 0; t < 6; t++) {
    int c = lane + 32 * t;
    orow[c] = (_Float16)((v[t] - mean) * inv * g[c] + b[c]);
  }
}

// ---------------------------------------------------------------------------
// Tiled WMMA GEMM: Y[m,n] = sum_k A[m,k] * W[n,k]  (A f16 MxK, W f16 NxK)
// 256 threads = 8 waves; block tile 128x192 (A read once per 192 columns ->
// minimal HBM traffic, weights stay L2-resident); wave tile 32x96 = 2x6
// WMMA tiles. A staged through LDS (row stride 40 halves, conflict-free
// b128 reads), B fragments loaded directly as one v16h per lane.
// Epilogues (uniform branch):
//   1: q   -> (y+b)*scale, split heads -> qh[win,head,tok,32]
//   2: kv  -> +b, n<192 -> kh[win,head,tok,32]; else vT[win,head,d,64] (transposed)
//   3: proj-> +b, window-reverse + roll(+4,+4), add shortcut x, f32 out
//   4: fc1 -> +b, exact GELU, f16 out row-major
//   5: fc2 -> +b, add residual, f32 out
// ---------------------------------------------------------------------------
__global__ __launch_bounds__(256)
void gemm_wmma(const _Float16* __restrict__ A, const _Float16* __restrict__ Wt,
               const float* __restrict__ bias, int K, int Nn, int epi, float scale,
               _Float16* __restrict__ outH, _Float16* __restrict__ outH2,
               float* __restrict__ outF, const float* __restrict__ addF) {
  __shared__ __align__(32) _Float16 As[128 * 40];
  int lane = threadIdx.x & 31;
  int wv   = threadIdx.x >> 5;
  int wrow = wv >> 1, wcol = wv & 1;
  int l15  = lane & 15, hib = lane >> 4;
  int m0 = blockIdx.x * 128;
  int n0 = blockIdx.y * 192;
  int srow = threadIdx.x >> 1;
  int scol = (threadIdx.x & 1) * 16;
  int abase = hib * 16;         // A-frag K base within 32-block (contiguous K order, matches B)
  int kb    = hib * 16;         // B-frag K base (contiguous K order, matches A)
  v8f acc[2][6] = {};

  for (int k0 = 0; k0 < K; k0 += 32) {
    __syncthreads();
    {
      const _Float16* gp = A + (size_t)(m0 + srow) * K + k0 + scol;
      v16h t = *(const v16h*)gp;
      if (k0 + 32 < K) __builtin_prefetch(gp + 32, 0, 0);
      v8h lo, hi;
#pragma unroll
      for (int i = 0; i < 8; i++) { lo[i] = t[i]; hi[i] = t[i + 8]; }
      *(v8h*)(As + srow * 40 + scol)     = lo;
      *(v8h*)(As + srow * 40 + scol + 8) = hi;
    }
    __syncthreads();

    v16h bfr[6];
#pragma unroll
    for (int ni = 0; ni < 6; ni++) {
      int n = n0 + wcol * 96 + ni * 16 + l15;
      bfr[ni] = *(const v16h*)(Wt + (size_t)n * K + k0 + kb);
    }
#pragma unroll
    for (int mi = 0; mi < 2; mi++) {
      int m = wrow * 32 + mi * 16 + l15;
      v8h p0 = *(const v8h*)(As + m * 40 + abase);
      v8h p1 = *(const v8h*)(As + m * 40 + abase + 8);
      v16h a;
#pragma unroll
      for (int i = 0; i < 8; i++) { a[i] = p0[i]; a[i + 8] = p1[i]; }
#pragma unroll
      for (int ni = 0; ni < 6; ni++) {
        acc[mi][ni] = __builtin_amdgcn_wmma_f32_16x16x32_f16(
            false, a, false, bfr[ni], (short)0, acc[mi][ni], false, false);
      }
    }
  }

  float bn[6];
#pragma unroll
  for (int ni = 0; ni < 6; ni++) bn[ni] = bias[n0 + wcol * 96 + ni * 16 + l15];

#pragma unroll
  for (int mi = 0; mi < 2; mi++) {
#pragma unroll
    for (int ni = 0; ni < 6; ni++) {
      int n = n0 + wcol * 96 + ni * 16 + l15;
#pragma unroll
      for (int j = 0; j < 8; j++) {
        int m = m0 + wrow * 32 + mi * 16 + hib * 8 + j;   // ISA C/D layout
        float y = acc[mi][ni][j] + bn[ni];
        if (epi == 1) {                 // Q: scale + head split
          y *= scale;
          int head = n >> 5, d = n & 31;
          int win = m >> 6, tok = m & 63;
          outH[(((size_t)win * 6 + head) * 64 + tok) * 32 + d] = (_Float16)y;
        } else if (epi == 2) {          // K / V(transposed)
          int win = m >> 6, tok = m & 63;
          if (n < 192) {
            int head = n >> 5, d = n & 31;
            outH[(((size_t)win * 6 + head) * 64 + tok) * 32 + d] = (_Float16)y;
          } else {
            int n2 = n - 192;
            int head = n2 >> 5, d = n2 & 31;
            outH2[(((size_t)win * 6 + head) * 32 + d) * 64 + tok] = (_Float16)y;
          }
        } else if (epi == 3) {          // proj + un-shift + residual (f32)
          int win = m >> 6, tok = m & 63;
          int bb = win >> 10, wi = win & 1023;
          int hh = ((wi >> 5) << 3) + (tok >> 3);
          int ww = ((wi & 31) << 3) + (tok & 7);
          int dh = (hh + 4) & 255, dw = (ww + 4) & 255;   // roll(+SS)
          size_t idx = ((size_t)bb * 65536 + (size_t)dh * 256 + dw) * 192 + n;
          outF[idx] = y + addF[idx];
        } else if (epi == 4) {          // fc1 + exact GELU
          y = 0.5f * y * (1.f + erff(y * 0.70710678118654752f));
          outH[(size_t)m * Nn + n] = (_Float16)y;
        } else {                        // fc2 + residual -> final output
          size_t idx = (size_t)m * 192 + n;
          outF[idx] = y + addF[idx];
        }
      }
    }
  }
}

// ---------------------------------------------------------------------------
// Attention: one wave per (window, head). 6 waves / block, one window / block.
// QK^T: 16 WMMAs (K=32==HD). Softmax in registers (rows live in 16-lane
// halves of the C layout). P -> LDS (A-frag layout), P@V: 16 WMMAs (K=64).
// ---------------------------------------------------------------------------
__global__ __launch_bounds__(192)
void attn_kernel(const _Float16* __restrict__ qh, const _Float16* __restrict__ kh,
                 const _Float16* __restrict__ vT, const float* __restrict__ bf,
                 const float* __restrict__ mask, _Float16* __restrict__ outH) {
  __shared__ __align__(32) _Float16 P[6][64 * 72];
  int wd = blockIdx.x;
  int h  = threadIdx.x >> 5;
  int lane = threadIdx.x & 31;
  int l15 = lane & 15, hib = lane >> 4;
  int abase = hib * 16, kb2 = hib * 16;
  const size_t baseQK = ((size_t)wd * 6 + h) * 64 * 32;

  v16h qa[4], kbf[4];
#pragma unroll
  for (int ti = 0; ti < 4; ti++) {
    int m = ti * 16 + l15;
    v8h p0 = *(const v8h*)(qh + baseQK + (size_t)m * 32 + abase);
    v8h p1 = *(const v8h*)(qh + baseQK + (size_t)m * 32 + abase + 8);
#pragma unroll
    for (int i = 0; i < 8; i++) { qa[ti][i] = p0[i]; qa[ti][i + 8] = p1[i]; }
  }
#pragma unroll
  for (int tj = 0; tj < 4; tj++) {
    int nt = tj * 16 + l15;
    kbf[tj] = *(const v16h*)(kh + baseQK + (size_t)nt * 32 + kb2);
  }

  v8f s[4][4] = {};
#pragma unroll
  for (int ti = 0; ti < 4; ti++)
#pragma unroll
    for (int tj = 0; tj < 4; tj++)
      s[ti][tj] = __builtin_amdgcn_wmma_f32_16x16x32_f16(
          false, qa[ti], false, kbf[tj], (short)0, s[ti][tj], false, false);

  int wi = wd & 1023;
  const float* bh = bf + (size_t)h * 4096;
  const float* mh = mask + (size_t)wi * 4096;
#pragma unroll
  for (int ti = 0; ti < 4; ti++) {
#pragma unroll
    for (int j = 0; j < 8; j++) {
      int r = ti * 16 + hib * 8 + j;
      float mx = -3.0e38f;
#pragma unroll
      for (int tj = 0; tj < 4; tj++) {
        int c = tj * 16 + l15;
        float v = s[ti][tj][j] + bh[r * 64 + c] + mh[r * 64 + c];
        s[ti][tj][j] = v;
        mx = fmaxf(mx, v);
      }
#pragma unroll
      for (int msk = 1; msk < 16; msk <<= 1) mx = fmaxf(mx, __shfl_xor(mx, msk, 32));
      float sum = 0.f;
#pragma unroll
      for (int tj = 0; tj < 4; tj++) {
        float e = __expf(s[ti][tj][j] - mx);
        s[ti][tj][j] = e;
        sum += e;
      }
#pragma unroll
      for (int msk = 1; msk < 16; msk <<= 1) sum += __shfl_xor(sum, msk, 32);
      float inv = 1.f / sum;
#pragma unroll
      for (int tj = 0; tj < 4; tj++) {
        int c = tj * 16 + l15;
        P[h][r * 72 + c] = (_Float16)(s[ti][tj][j] * inv);
      }
    }
  }

  // out = P @ V   (M=64, N=32, K=64 -> 2 k-steps)
  const size_t baseV = ((size_t)wd * 6 + h) * 32 * 64;
  v8f o[4][2] = {};
#pragma unroll
  for (int ks = 0; ks < 2; ks++) {
    v16h vb[2];
#pragma unroll
    for (int ni = 0; ni < 2; ni++) {
      int d = ni * 16 + l15;
      vb[ni] = *(const v16h*)(vT + baseV + (size_t)d * 64 + ks * 32 + kb2);
    }
#pragma unroll
    for (int ti = 0; ti < 4; ti++) {
      const _Float16* pr = &P[h][(ti * 16 + l15) * 72 + ks * 32 + abase];
      v8h p0 = *(const v8h*)pr;
      v8h p1 = *(const v8h*)(pr + 8);
      v16h a;
#pragma unroll
      for (int i = 0; i < 8; i++) { a[i] = p0[i]; a[i + 8] = p1[i]; }
#pragma unroll
      for (int ni = 0; ni < 2; ni++)
        o[ti][ni] = __builtin_amdgcn_wmma_f32_16x16x32_f16(
            false, a, false, vb[ni], (short)0, o[ti][ni], false, false);
    }
  }

#pragma unroll
  for (int ti = 0; ti < 4; ti++)
#pragma unroll
    for (int ni = 0; ni < 2; ni++)
#pragma unroll
      for (int j = 0; j < 8; j++) {
        int m = ti * 16 + hib * 8 + j;
        int d = ni * 16 + l15;
        outH[((size_t)wd * 64 + m) * 192 + h * 32 + d] = (_Float16)o[ti][ni][j];
      }
}

// ---------------------------------------------------------------------------
// launch
// ---------------------------------------------------------------------------
extern "C" void kernel_launch(void* const* d_in, const int* in_sizes, int n_in,
                              void* d_out, int out_size, void* d_ws, size_t ws_size,
                              hipStream_t stream) {
  (void)in_sizes; (void)n_in; (void)out_size; (void)ws_size;
  const float* x   = (const float*)d_in[0];
  const float* cx  = (const float*)d_in[1];
  const int*   rpi = (const int*)d_in[2];
  const float* msk = (const float*)d_in[3];
  const float* n1g = (const float*)d_in[4];
  const float* n1b = (const float*)d_in[5];
  const float* ncg = (const float*)d_in[6];
  const float* ncb = (const float*)d_in[7];
  const float* qw  = (const float*)d_in[8];
  const float* qb  = (const float*)d_in[9];
  const float* kvw = (const float*)d_in[10];
  const float* kvb = (const float*)d_in[11];
  const float* rpb = (const float*)d_in[12];
  const float* pw  = (const float*)d_in[13];
  const float* pb  = (const float*)d_in[14];
  const float* n2g = (const float*)d_in[15];
  const float* n2b = (const float*)d_in[16];
  const float* f1w = (const float*)d_in[17];
  const float* f1b = (const float*)d_in[18];
  const float* f2w = (const float*)d_in[19];
  const float* f2b = (const float*)d_in[20];
  float* out = (float*)d_out;

  char* ws = (char*)d_ws;
  _Float16* wq    = (_Float16*)(ws + 0);
  _Float16* wkv   = (_Float16*)(ws + 73728);
  _Float16* wpj   = (_Float16*)(ws + 221184);
  _Float16* wf1   = (_Float16*)(ws + 294912);
  _Float16* wf2   = (_Float16*)(ws + 589824);
  float*    bfull = (float*)   (ws + 884736);
  _Float16* xwh   = (_Float16*)(ws + 983040);      // 100.66 MB
  _Float16* cwh   = (_Float16*)(ws + 101646336);
  _Float16* q_h   = (_Float16*)(ws + 202309632);
  _Float16* k_h   = (_Float16*)(ws + 302972928);
  _Float16* vT_h  = (_Float16*)(ws + 403636224);
  _Float16* h1_h  = (_Float16*)(ws + 504299520);   // 402.65 MB
  _Float16* ao_h  = xwh;                            // reuse (xw dead after q-gemm)
  float*    x1    = (float*)   (ws + 101646336);    // reuse cw+q span (201.3 MB)
  _Float16* h2_h  = k_h;                            // reuse (k dead after attn)

  f2h_kernel<<<(36864  + 255) / 256, 256, 0, stream>>>(qw,  wq,  36864);
  f2h_kernel<<<(73728  + 255) / 256, 256, 0, stream>>>(kvw, wkv, 73728);
  f2h_kernel<<<(36864  + 255) / 256, 256, 0, stream>>>(pw,  wpj, 36864);
  f2h_kernel<<<(147456 + 255) / 256, 256, 0, stream>>>(f1w, wf1, 147456);
  f2h_kernel<<<(147456 + 255) / 256, 256, 0, stream>>>(f2w, wf2, 147456);
  biasexp_kernel<<<(24576 + 255) / 256, 256, 0, stream>>>(rpi, rpb, bfull);

  ln_kernel<<<32768, 256, 0, stream>>>(x,  n1g, n1b, xwh, 0);
  ln_kernel<<<32768, 256, 0, stream>>>(cx, ncg, ncb, cwh, 0);

  // q = LN(x)W_q^T * scale ; kv = LN(cx)W_kv^T
  gemm_wmma<<<dim3(2048, 1), 256, 0, stream>>>(xwh, wq,  qb,  192, 192, 1,
                                               0.17677669529663687f, q_h, nullptr, nullptr, nullptr);
  gemm_wmma<<<dim3(2048, 2), 256, 0, stream>>>(cwh, wkv, kvb, 192, 384, 2,
                                               1.f, k_h, vT_h, nullptr, nullptr);

  attn_kernel<<<4096, 192, 0, stream>>>(q_h, k_h, vT_h, bfull, msk, ao_h);

  // x1 = x + unshift(proj(attn_out))
  gemm_wmma<<<dim3(2048, 1), 256, 0, stream>>>(ao_h, wpj, pb, 192, 192, 3,
                                               1.f, nullptr, nullptr, x1, x);

  ln_kernel<<<32768, 256, 0, stream>>>(x1, n2g, n2b, h2_h, 1);

  // h1 = gelu(LN2(x1) W1^T + b1) ; out = x1 + h1 W2^T + b2
  gemm_wmma<<<dim3(2048, 4), 256, 0, stream>>>(h2_h, wf1, f1b, 192, 768, 4,
                                               1.f, h1_h, nullptr, nullptr, nullptr);
  gemm_wmma<<<dim3(2048, 1), 256, 0, stream>>>(h1_h, wf2, f2b, 768, 192, 5,
                                               1.f, nullptr, nullptr, out, x1);
}
